// Qwen3TTSAttentionStandalone_17583596109938
// MI455X (gfx1250) — matrix operand order repaired and checked by reading, not compiled, not run on hardware
//
#include <hip/hip_runtime.h>
#include <hip/hip_bf16.h>

// ---------------- constants (match reference) ----------------
#define BB   4
#define SS   1024
#define HID  1024
#define NH   16
#define NKV  8
#define HD   64
#define TOK  (BB*SS)          // 4096
#define NQKV 2048             // Wq(1024)+Wk(512)+Wv(512) output cols
#define EPSV 1e-6f

typedef __bf16 bf16_t;
typedef __attribute__((ext_vector_type(16))) __bf16 v16bf;
typedef __attribute__((ext_vector_type(8)))  __bf16 v8bf;
typedef __attribute__((ext_vector_type(4)))  __bf16 v4bf;
typedef __attribute__((ext_vector_type(8)))  float  v8f;

#define WMMA_BF16(a, b, c) \
  __builtin_amdgcn_wmma_f32_16x16x32_bf16(false, (a), false, (b), (short)0, (c), false, false)

// ---------------- fragment helpers (wave32; contiguous k order chosen by this file, not the ISA table order) ----------------
// A (16x32 bf16): lane L holds row M=L%16; element e -> k = 16*(L/16) + e.
// => one contiguous 16-half run at (base + 16*laneHi).
__device__ __forceinline__ v16bf load_frag_a(const bf16_t* p) {
  v8bf lo = *(const v8bf*)(p);
  v8bf hi = *(const v8bf*)(p + 8);
  v16bf r;
#pragma unroll
  for (int i = 0; i < 8; ++i) { r[i] = lo[i]; r[8 + i] = hi[i]; }
  return r;
}
// B (32x16 bf16): lane L holds col N=L%16; element e -> k = 16*(L/16) + e.
// => one contiguous 16-half (32B-aligned) run.
__device__ __forceinline__ v16bf load_frag_b(const bf16_t* p) {
  return *(const v16bf*)(p);
}
__device__ __forceinline__ v8f zero8() {
  v8f z;
#pragma unroll
  for (int i = 0; i < 8; ++i) z[i] = 0.f;
  return z;
}
__device__ __forceinline__ float wave_sum(float v) {
#pragma unroll
  for (int m = 16; m >= 1; m >>= 1) v += __shfl_xor(v, m, 32);
  return v;
}
__device__ __forceinline__ float wave_max(float v) {
#pragma unroll
  for (int m = 16; m >= 1; m >>= 1) v = fmaxf(v, __shfl_xor(v, m, 32));
  return v;
}

// ---------------- kernel 1: f32 -> bf16 convert (x4 vectorized) ----------------
__global__ void cvt_f32_bf16(const float* __restrict__ src, bf16_t* __restrict__ dst, int n4) {
  int i = blockIdx.x * blockDim.x + threadIdx.x;
  if (i < n4) {
    float4 v = ((const float4*)src)[i];
    v4bf r;
    r[0] = (bf16_t)v.x; r[1] = (bf16_t)v.y; r[2] = (bf16_t)v.z; r[3] = (bf16_t)v.w;
    *(v4bf*)(dst + 4 * (size_t)i) = r;
  }
}

// ---------------- kernel 2: bf16 WMMA GEMM, 16x64 per wave, ping-pong pipelined ----------------
// C[M,N] (f32, row-major) = A[M,K] (bf16 row-major) * W[N,K]^T (bf16 row-major)
// block = 4 waves (128 thr) covering 64 rows x 64 cols. K unrolled by 64 with two named
// fragment sets (no rotation copies): set-1 loads issue before set-0 WMMAs; set-0 refills
// for the next iteration between the two WMMA groups -> loads overlap matrix ops.
__global__ __launch_bounds__(128, 1)
void gemm_bf16(const bf16_t* __restrict__ A, const bf16_t* __restrict__ Bw,
               float* __restrict__ C, int M, int N, int K) {
  const int lane   = threadIdx.x & 31;
  const int laneLo = lane & 15;
  const int laneHi = lane >> 4;
  const int wave   = threadIdx.x >> 5;
  const int rbase  = blockIdx.y * 64 + wave * 16;
  const int nt0    = blockIdx.x * 4;

  const bf16_t* arow = A + (size_t)(rbase + laneLo) * K + 16 * laneHi;
  const bf16_t* brow[4];
#pragma unroll
  for (int t = 0; t < 4; ++t)
    brow[t] = Bw + (size_t)((nt0 + t) * 16 + laneLo) * K + 16 * laneHi;

  v8f acc[4];
#pragma unroll
  for (int t = 0; t < 4; ++t) acc[t] = zero8();

  // prologue: set 0 (k = 0..31)
  v16bf a0 = load_frag_a(arow);
  v16bf b0[4];
#pragma unroll
  for (int t = 0; t < 4; ++t) b0[t] = load_frag_b(brow[t]);

  for (int kk = 0; kk < K; kk += 64) {       // K is a multiple of 64
    // set 1 loads (k = kk+32) issued ahead of set-0 WMMAs
    v16bf a1 = load_frag_a(arow + kk + 32);
    v16bf b1[4];
#pragma unroll
    for (int t = 0; t < 4; ++t) b1[t] = load_frag_b(brow[t] + kk + 32);
    if (kk + 512 < K) __builtin_prefetch(arow + kk + 512, 0, 1);   // global_prefetch

#pragma unroll
    for (int t = 0; t < 4; ++t) acc[t] = WMMA_BF16(a0, b0[t], acc[t]);

    // refill set 0 for next iteration while set-1 WMMAs run
    if (kk + 64 < K) {
      a0 = load_frag_a(arow + kk + 64);
#pragma unroll
      for (int t = 0; t < 4; ++t) b0[t] = load_frag_b(brow[t] + kk + 64);
    }
#pragma unroll
    for (int t = 0; t < 4; ++t) acc[t] = WMMA_BF16(a1, b1[t], acc[t]);
  }

  // D/C layout: VGPR r -> row = r + 8*laneHi, col = laneLo (per 16-col tile)
#pragma unroll
  for (int t = 0; t < 4; ++t) {
#pragma unroll
    for (int r = 0; r < 8; ++r) {
      C[(size_t)(rbase + r + 8 * laneHi) * N + (nt0 + t) * 16 + laneLo] = acc[t][r];
    }
  }
}

// ---------------- kernel 3: RMSNorm + RoPE + head split (+ V transpose) ----------------
// grid: (32 jobs, TOK tokens); jobs 0..15 = Q heads, 16..23 = K heads, 24..31 = V heads.
__global__ void norm_rope(const float* __restrict__ qkv,
                          const float* __restrict__ cosp, const float* __restrict__ sinp,
                          const float* __restrict__ qw, const float* __restrict__ kw,
                          bf16_t* __restrict__ qbf, bf16_t* __restrict__ kbf,
                          bf16_t* __restrict__ vT) {
  const int lane = threadIdx.x;            // 0..31, owns dims (lane, lane+32)
  const int job  = blockIdx.x;
  const int tok  = blockIdx.y;
  const int b = tok >> 10, s = tok & 1023;

  if (job < 24) {  // Q or K: RMSNorm + RoPE
    const bool isQ = job < 16;
    const int  h   = isQ ? job : (job - 16);
    const size_t base = (size_t)tok * NQKV + (isQ ? 0 : 1024) + h * HD;
    float x1 = qkv[base + lane], x2 = qkv[base + lane + 32];
    float ssum = wave_sum(x1 * x1 + x2 * x2);
    float inv  = rsqrtf(ssum * (1.f / 64.f) + EPSV);
    const float* w = isQ ? qw : kw;
    float y1 = x1 * inv * w[lane];
    float y2 = x2 * inv * w[lane + 32];
    size_t cb = ((size_t)(b * SS + s)) * HD;
    float c1 = cosp[cb + lane], c2 = cosp[cb + lane + 32];
    float s1 = sinp[cb + lane], s2 = sinp[cb + lane + 32];
    float o1 = y1 * c1 - y2 * s1;     // rotate_half: first half gets -x2*sin
    float o2 = y2 * c2 + y1 * s2;     // second half gets +x1*sin
    size_t ob = isQ ? (((size_t)(b * NH + h) * SS + s) * HD)
                    : (((size_t)(b * NKV + h) * SS + s) * HD);
    bf16_t* dst = isQ ? qbf : kbf;
    dst[ob + lane]      = (bf16_t)o1;
    dst[ob + lane + 32] = (bf16_t)o2;
  } else {         // V: convert + transpose (d-major for WMMA B operand)
    const int kv = job - 24;
    const size_t base = (size_t)tok * NQKV + 1536 + kv * HD;
    float x1 = qkv[base + lane], x2 = qkv[base + lane + 32];
    size_t vb = (size_t)(b * NKV + kv) * HD;
    vT[(vb + lane) * SS + s]        = (bf16_t)x1;
    vT[(vb + lane + 32) * SS + s]   = (bf16_t)x2;
  }
}

// ---------------- kernel 4: attention (scores -> softmax -> attn_weights -> P*V) ----------------
// grid: (B*NH, S/16); one wave per block; 64KB dynamic LDS holds 16x1024 f32 scores.
__global__ __launch_bounds__(32, 1)
void attn_kernel(const bf16_t* __restrict__ qbf, const bf16_t* __restrict__ kbf,
                 const bf16_t* __restrict__ vT,
                 float* __restrict__ attn, bf16_t* __restrict__ ctxbf) {
  extern __shared__ float smem[];                 // [16][1024]
  const int lane   = threadIdx.x;
  const int laneLo = lane & 15;
  const int laneHi = lane >> 4;
  const int bh = blockIdx.x, b = bh >> 4, h = bh & 15, kv = h >> 1;   // GQA: 2 q-heads / kv-head
  const int qt = blockIdx.y, qrow0 = qt * 16;

  const bf16_t* qbase = qbf + ((size_t)(b * NH + h) * SS + qrow0 + laneLo) * HD;
  const bf16_t* kptr  = kbf + ((size_t)(b * NKV + kv) * SS + laneLo) * HD + 16 * laneHi;

  // score-tile epilogue: scale + causal mask -> LDS
  auto score_epi = [&](const v8f& acc, int ct) {
#pragma unroll
    for (int r = 0; r < 8; ++r) {
      int lr  = r + 8 * laneHi;
      int row = qrow0 + lr;
      int col = ct * 16 + laneLo;
      float sc = acc[r] * 0.125f;                 // HD^-0.5
      if (col > row) sc = -1000000000.0f;         // causal mask (== attention_mask values)
      smem[lr * SS + col] = sc;
    }
  };

  // ---- phase 1: scores over causal tiles only; Q fragments loop-invariant;
  //      K-tiles processed in ping-pong pairs (no rotation copies) ----
  v16bf aq0 = load_frag_a(qbase + 16 * laneHi);
  v16bf aq1 = load_frag_a(qbase + 32 + 16 * laneHi);
  v16bf c0k0 = load_frag_b(kptr);
  v16bf c0k1 = load_frag_b(kptr + 32);
  for (int ct = 0; ct <= qt; ct += 2) {
    const bool has2 = (ct + 1 <= qt);
    v16bf c1k0, c1k1;
    if (has2) {
      const bf16_t* np = kptr + (size_t)(ct + 1) * 16 * HD;
      c1k0 = load_frag_b(np);
      c1k1 = load_frag_b(np + 32);
    }
    v8f acc = zero8();
    acc = WMMA_BF16(aq0, c0k0, acc);
    acc = WMMA_BF16(aq1, c0k1, acc);
    if (ct + 2 <= qt) {
      const bf16_t* np2 = kptr + (size_t)(ct + 2) * 16 * HD;
      c0k0 = load_frag_b(np2);
      c0k1 = load_frag_b(np2 + 32);
    }
    score_epi(acc, ct);
    if (has2) {
      v8f acc2 = zero8();
      acc2 = WMMA_BF16(aq0, c1k0, acc2);
      acc2 = WMMA_BF16(aq1, c1k1, acc2);
      score_epi(acc2, ct + 1);
    }
  }
  __syncthreads();

  // ---- phase 2: f32 softmax + write attn_weights (zeros beyond causal) ----
  const int ncomp = (qt + 1) * 16;
  for (int r = 0; r < 16; ++r) {
    const int row = qrow0 + r;
    float mx = -3.4e38f;
    for (int c = lane; c < ncomp; c += 32) mx = fmaxf(mx, smem[r * SS + c]);
    mx = wave_max(mx);
    float sum = 0.f;
    for (int c = lane; c < ncomp; c += 32) {
      float e = __expf(smem[r * SS + c] - mx);
      smem[r * SS + c] = e;
      sum += e;
    }
    sum = wave_sum(sum);
    float inv = 1.f / sum;
    float* arow = attn + ((size_t)bh * SS + row) * SS;
    for (int c = lane; c < ncomp; c += 32) {
      float p = smem[r * SS + c] * inv;
      smem[r * SS + c] = p;
      arow[c] = p;
    }
    for (int c = ncomp + lane; c < SS; c += 32) arow[c] = 0.f;
  }
  // pad LDS to a multiple of the 32-wide WMMA K-step (at most 16 extra columns)
  const int kend = (ncomp + 31) & ~31;
  if (kend > ncomp) {
    for (int idx = lane; idx < 256; idx += 32) {
      int r = idx >> 4, c = ncomp + (idx & 15);
      smem[r * SS + c] = 0.f;
    }
  }
  __syncthreads();

  // ---- phase 3: ctx(16x64) = P(16xkend) * V(kend x 64); K-steps in ping-pong pairs ----
  v8f acc[4];
#pragma unroll
  for (int t = 0; t < 4; ++t) acc[t] = zero8();
  const bf16_t* vptr = vT + ((size_t)(b * NKV + kv) * HD + laneLo) * SS + 16 * laneHi;

  // P fragment from LDS (f32 -> bf16 in registers), A layout
  auto load_p = [&](int kk) {
    const float* pr = smem + laneLo * SS + kk + 16 * laneHi;
    v16bf a;
#pragma unroll
    for (int i = 0; i < 8; ++i) { a[i] = (bf16_t)pr[i]; a[8 + i] = (bf16_t)pr[8 + i]; }
    return a;
  };

  v16bf bv0[4];
#pragma unroll
  for (int t = 0; t < 4; ++t) bv0[t] = load_frag_b(vptr + (size_t)t * 16 * SS);

  for (int kk = 0; kk < kend; kk += 64) {    // kend is a multiple of 32
    const bool has2 = (kk + 32 < kend);
    v16bf bv1[4];
    if (has2) {
#pragma unroll
      for (int t = 0; t < 4; ++t) bv1[t] = load_frag_b(vptr + (size_t)t * 16 * SS + kk + 32);
    }
    v16bf a = load_p(kk);
#pragma unroll
    for (int t = 0; t < 4; ++t) acc[t] = WMMA_BF16(a, bv0[t], acc[t]);
    if (kk + 64 < kend) {
#pragma unroll
      for (int t = 0; t < 4; ++t) bv0[t] = load_frag_b(vptr + (size_t)t * 16 * SS + kk + 64);
    }
    if (has2) {
      v16bf a2 = load_p(kk + 32);
#pragma unroll
      for (int t = 0; t < 4; ++t) acc[t] = WMMA_BF16(a2, bv1[t], acc[t]);
    }
  }

  // ctx -> [b, s, h*64 + d] bf16 for the output projection
#pragma unroll
  for (int t = 0; t < 4; ++t) {
#pragma unroll
    for (int r = 0; r < 8; ++r) {
      int row = qrow0 + r + 8 * laneHi;
      ctxbf[((size_t)(b * SS + row)) * (NH * HD) + h * HD + t * 16 + laneLo] = (bf16_t)acc[t][r];
    }
  }
}

// ---------------- host-side orchestration ----------------
extern "C" void kernel_launch(void* const* d_in, const int* in_sizes, int n_in,
                              void* d_out, int out_size, void* d_ws, size_t ws_size,
                              hipStream_t stream) {
  (void)in_sizes; (void)n_in; (void)out_size; (void)ws_size;
  const float* hs   = (const float*)d_in[0];
  const float* cosp = (const float*)d_in[1];
  const float* sinp = (const float*)d_in[2];
  // d_in[3] = attention_mask: causal 0/-1e9, reproduced analytically in-kernel.
  const float* Wq = (const float*)d_in[4];
  const float* Wk = (const float*)d_in[5];
  const float* Wv = (const float*)d_in[6];
  const float* Wo = (const float*)d_in[7];
  const float* qw = (const float*)d_in[8];
  const float* kw = (const float*)d_in[9];

  // workspace partitioning (256B aligned)
  char* ws = (char*)d_ws;
  size_t off = 0;
  auto take = [&](size_t bytes) {
    char* p = ws + off;
    off += (bytes + 255) & ~(size_t)255;
    return p;
  };
  bf16_t* hidden_bf = (bf16_t*)take((size_t)TOK * HID * 2);          // 8 MB
  bf16_t* wqkv_bf   = (bf16_t*)take((size_t)NQKV * HID * 2);         // 4 MB
  bf16_t* wo_bf     = (bf16_t*)take((size_t)HID * HID * 2);          // 2 MB
  float*  qkv_f32   = (float*) take((size_t)TOK * NQKV * 4);         // 32 MB
  bf16_t* q_bf      = (bf16_t*)take((size_t)BB * NH  * SS * HD * 2); // 8 MB
  bf16_t* k_bf      = (bf16_t*)take((size_t)BB * NKV * SS * HD * 2); // 4 MB
  bf16_t* vT_bf     = (bf16_t*)take((size_t)BB * NKV * HD * SS * 2); // 4 MB
  bf16_t* ctx_bf    = (bf16_t*)take((size_t)TOK * NH * HD * 2);      // 8 MB

  float* out_proj  = (float*)d_out;                       // (B,S,HID)
  float* attn_out  = (float*)d_out + (size_t)TOK * HID;   // (B,NH,S,S)

  // 1) conversions to bf16 (weights keep (N,K) row-major => contiguous WMMA B loads)
  {
    int n4;
    n4 = TOK * HID / 4;  cvt_f32_bf16<<<(n4 + 255) / 256, 256, 0, stream>>>(hs, hidden_bf, n4);
    n4 = 1024 * HID / 4; cvt_f32_bf16<<<(n4 + 255) / 256, 256, 0, stream>>>(Wq, wqkv_bf, n4);
    n4 = 512 * HID / 4;  cvt_f32_bf16<<<(n4 + 255) / 256, 256, 0, stream>>>(Wk, wqkv_bf + (size_t)1024 * HID, n4);
    n4 = 512 * HID / 4;  cvt_f32_bf16<<<(n4 + 255) / 256, 256, 0, stream>>>(Wv, wqkv_bf + (size_t)1536 * HID, n4);
    n4 = HID * HID / 4;  cvt_f32_bf16<<<(n4 + 255) / 256, 256, 0, stream>>>(Wo, wo_bf, n4);
  }

  // 2) fused QKV projection: [4096 x 1024] x [2048 x 1024]^T -> [4096 x 2048]
  gemm_bf16<<<dim3(NQKV / 64, TOK / 64), 128, 0, stream>>>(hidden_bf, wqkv_bf, qkv_f32,
                                                           TOK, NQKV, HID);

  // 3) RMSNorm + RoPE + head split (+ V transpose)
  norm_rope<<<dim3(32, TOK), 32, 0, stream>>>(qkv_f32, cosp, sinp, qw, kw, q_bf, k_bf, vT_bf);

  // 4) attention: one wave per (head, 16-row q tile); 64KB LDS score buffer
  attn_kernel<<<dim3(BB * NH, SS / 16), 32, 16 * SS * sizeof(float), stream>>>(
      q_bf, k_bf, vT_bf, attn_out, ctx_bf);

  // 5) output projection: [4096 x 1024] x [1024 x 1024]^T -> d_out
  gemm_bf16<<<dim3(HID / 64, TOK / 64), 128, 0, stream>>>(ctx_bf, wo_bf, out_proj,
                                                          TOK, HID, HID);
}
